// WindowAttention_53506702574102
// MI455X (gfx1250) — matrix operand order repaired and checked by reading, not compiled, not run on hardware
//
#include <hip/hip_runtime.h>
#include <hip/hip_bf16.h>

typedef __attribute__((ext_vector_type(16))) _Float16 v16h;
typedef __attribute__((ext_vector_type(8)))  _Float16 v8h;
typedef __attribute__((ext_vector_type(8)))  float    v8f;
typedef __attribute__((ext_vector_type(4)))  float    v4f;

#define NTOK 49
#define NPAD 64
#define NH   8
#define DH   32
#define DIMC 256
#define NWIN 2048

static __device__ inline v16h mk16(v8h lo, v8h hi) {
  union { v16h v; v8h h[2]; } u;
  u.h[0] = lo; u.h[1] = hi;
  return u.v;
}

// A-fragment (16x32 f16, M x K): lane row = lrow; lo/hi lane half selects K
// phase per ISA 7.12.2 (k0..7 & k16..23 vs k8..15 & k24..31).
static __device__ inline v16h frag_a(const _Float16* base, int row, int stride,
                                     int k0, int lhi) {
  const _Float16* p = base + row * stride + k0 + lhi * 8;
  return mk16(*(const v8h*)p, *(const v8h*)(p + 16));
}

// B-fragment (32x16 f16, K x N) from a row-major [N][K] matrix: lane col = n;
// lo lanes hold k0..7 & k16..23, hi lanes k8..15 & k24..31 (matches frag_a).
static __device__ inline v16h frag_b(const _Float16* base, int col, int stride,
                                     int k0, int lhi) {
  const _Float16* p = base + col * stride + k0 + lhi * 8;
  return mk16(*(const v8h*)p, *(const v8h*)(p + 16));
}

static __device__ inline v8f wmma16(v16h A, v16h B, v8f C) {
  return __builtin_amdgcn_wmma_f32_16x16x32_f16(false, A, false, B, (short)0, C,
                                                false, false);
}

// ---------------------------------------------------------------------------
// Tiny prepass: fp32 weights -> f16 scratch (then L2-resident for all blocks).
// ---------------------------------------------------------------------------
__global__ __launch_bounds__(256) void k_convert_weights(
    const float* __restrict__ qkvw, const float* __restrict__ projw,
    _Float16* __restrict__ wq, _Float16* __restrict__ wp) {
  int i = blockIdx.x * 256 + threadIdx.x;
  if (i < 3 * DIMC * DIMC) wq[i] = (_Float16)qkvw[i];
  if (i < DIMC * DIMC)     wp[i] = (_Float16)projw[i];
}

// ---------------------------------------------------------------------------
// Fully fused window attention: one block per window, everything in LDS.
//   ph0: stage x_b (f32->f16, pad 49->64 rows with zeros)
//   ph1: qkv GEMM (K=256) -> q (scaled) / k / v^T per-head LDS layouts
//   ph2/3 (x2 batches of 4 heads, 2 waves per head):
//        S = q k^T (1 WMMA step, K=32) -> f16 scores in SP
//        in-place softmax with rpb gather + window mask (bias recomputed
//        per pass so no f32 score buffer is needed)
//        O = P v (2 WMMA steps over padded K=64) -> AO tile (reuses x region)
//   ph4: proj GEMM (K=256) -> fp32 out + bias
// ---------------------------------------------------------------------------
__global__ __launch_bounds__(256) void k_fused_window_attn(
    const float* __restrict__ x, const float* __restrict__ mask,
    const _Float16* __restrict__ wq, const float* __restrict__ qkvb,
    const _Float16* __restrict__ wp, const float* __restrict__ pb,
    const float* __restrict__ rpb, const int* __restrict__ relidx,
    float* __restrict__ out) {
  __shared__ alignas(32) _Float16 XA[NPAD * DIMC];   // 32 KB: x, later attn-out
  __shared__ alignas(32) _Float16 QKV[3 * NH * NPAD * DH]; // 96 KB
  __shared__ alignas(32) _Float16 SP[4 * NPAD * NPAD];     // 32 KB: scores/probs
  _Float16* Q  = QKV;                    // [8][64][32], pre-scaled
  _Float16* K  = QKV + NH * NPAD * DH;   // [8][64][32]
  _Float16* VT = QKV + 2 * NH * NPAD * DH; // [8][32][64]

  const int b = blockIdx.x;
  const int tid = threadIdx.x;
  const int wave = tid >> 5, lane = tid & 31;
  const int lrow = lane & 15, lhi = lane >> 4;

  // ---- ph0: stage x_b into XA as f16, zero the 15 padded rows -------------
  for (int c = tid; c < NPAD * (DIMC / 8); c += 256) {
    int row = c >> 5, colg = (c & 31) * 8;
    v8h hc;
    if (row < NTOK) {
      const float* src = x + ((size_t)b * NTOK + row) * DIMC + colg;
      v4f s0 = *(const v4f*)src, s1 = *(const v4f*)(src + 4);
#pragma unroll
      for (int j = 0; j < 4; ++j) { hc[j] = (_Float16)s0[j]; hc[4 + j] = (_Float16)s1[j]; }
    } else {
#pragma unroll
      for (int j = 0; j < 8; ++j) hc[j] = (_Float16)0.f;
    }
    *(v8h*)&XA[row * DIMC + colg] = hc;
  }
  __syncthreads();

  // ---- ph1: qkv = XA @ wq^T (+bias), scattered into Q/K/VT ----------------
  {
    const int mt = wave >> 1;      // wave's 16-row tile
    const int np = wave & 1;       // n-tile parity (2 waves share an m-tile)
    const float scale = 0.17677669529663687f;  // Dh^-0.5
    for (int g = 0; g < 6; ++g) {  // 24 n-tiles per wave, groups of 4
      v8f acc[4] = {};
      for (int ks = 0; ks < DIMC; ks += 32) {
        v16h A = frag_a(XA, mt * 16 + lrow, DIMC, ks, lhi);
#pragma unroll
        for (int i = 0; i < 4; ++i) {
          int nt = np + 2 * (g * 4 + i);
          v16h B = frag_b(wq, nt * 16 + lrow, DIMC, ks, lhi);  // global, L2-hot
          acc[i] = wmma16(A, B, acc[i]);
        }
      }
#pragma unroll
      for (int i = 0; i < 4; ++i) {
        int nt = np + 2 * (g * 4 + i);
        int gc = nt * 16 + lrow;           // 0..767
        int which = gc >> 8;               // 0=q 1=k 2=v
        int h = (gc >> 5) & 7;
        int d = gc & 31;
        float bias = qkvb[gc];
#pragma unroll
        for (int r = 0; r < 8; ++r) {
          int t = mt * 16 + lhi * 8 + r;   // token row (incl. padded)
          float val = acc[i][r] + bias;
          if (which == 0)      Q[h * (NPAD * DH) + t * DH + d] = (_Float16)(val * scale);
          else if (which == 1) K[h * (NPAD * DH) + t * DH + d] = (_Float16)val;
          else                 VT[h * (DH * NPAD) + d * NPAD + t] = (_Float16)val;
        }
      }
    }
  }
  __syncthreads();

  // ---- ph2/ph3: attention, 2 batches of 4 heads, wave pair per head -------
  for (int batch = 0; batch < 2; ++batch) {
    const int hp = wave >> 1;              // head-in-batch 0..3
    const int h  = batch * 4 + hp;
    const _Float16* Qh = Q  + h * (NPAD * DH);
    const _Float16* Kh = K  + h * (NPAD * DH);
    const _Float16* Vh = VT + h * (DH * NPAD);
    _Float16* Sh = SP + hp * (NPAD * NPAD);

    // scores S = q k^T : wave handles 2 of the 4 m-tiles
#pragma unroll
    for (int mi = 0; mi < 2; ++mi) {
      int mt = (wave & 1) * 2 + mi;
      v16h A = frag_a(Qh, mt * 16 + lrow, DH, 0, lhi);
#pragma unroll
      for (int nt = 0; nt < 4; ++nt) {
        v16h B = frag_b(Kh, nt * 16 + lrow, DH, 0, lhi);
        v8f acc = {};
        acc = wmma16(A, B, acc);
#pragma unroll
        for (int r = 0; r < 8; ++r)
          Sh[(mt * 16 + lhi * 8 + r) * NPAD + nt * 16 + lrow] = (_Float16)acc[r];
      }
    }
    __syncthreads();

    // in-place softmax over f16 scores; one row per lane (2 waves = 64 rows)
    {
      int row = (wave & 1) * 32 + lane;
      _Float16* Prow = Sh + row * NPAD;
      if (row < NTOK) {
        const float* mrow = mask + ((size_t)(b & 63) * NTOK + row) * NTOK;
        const int*   rrow = relidx + row * NTOK;
        float mx = -1e30f;
        for (int j = 0; j < NTOK; ++j) {
          float v = (float)Prow[j] + rpb[rrow[j] * NH + h] + mrow[j];
          mx = fmaxf(mx, v);
        }
        float s = 0.f;
        for (int j = 0; j < NTOK; ++j) {
          float v = (float)Prow[j] + rpb[rrow[j] * NH + h] + mrow[j];
          float e = __expf(v - mx);
          s += e;
          Prow[j] = (_Float16)e;
        }
        float inv = 1.f / s;
        for (int j = 0; j < NTOK; ++j) Prow[j] = (_Float16)((float)Prow[j] * inv);
        for (int j = NTOK; j < NPAD; ++j) Prow[j] = (_Float16)0.f;
      } else {
        for (int j = 0; j < NPAD; ++j) Prow[j] = (_Float16)0.f;
      }
    }
    __syncthreads();

    // O = P v^T-layout : write into AO tile (reuses XA region; x is dead)
#pragma unroll
    for (int mi = 0; mi < 2; ++mi) {
      int mt = (wave & 1) * 2 + mi;
#pragma unroll
      for (int dt = 0; dt < 2; ++dt) {
        v8f acc = {};
#pragma unroll
        for (int ks = 0; ks < 2; ++ks) {
          v16h A = frag_a(Sh, mt * 16 + lrow, NPAD, ks * 32, lhi);
          v16h B = frag_b(Vh, dt * 16 + lrow, NPAD, ks * 32, lhi);
          acc = wmma16(A, B, acc);
        }
#pragma unroll
        for (int r = 0; r < 8; ++r) {
          int t = mt * 16 + lhi * 8 + r;
          XA[t * DIMC + h * DH + dt * 16 + lrow] = (_Float16)acc[r];
        }
      }
    }
    __syncthreads();
  }

  // ---- ph4: out = AO @ wp^T + bias (fp32 to HBM) --------------------------
  {
    const int mt = wave >> 1;
    const int np = wave & 1;
    for (int g = 0; g < 2; ++g) {   // 8 n-tiles per wave, groups of 4
      v8f acc[4] = {};
      for (int ks = 0; ks < DIMC; ks += 32) {
        v16h A = frag_a(XA, mt * 16 + lrow, DIMC, ks, lhi);
#pragma unroll
        for (int i = 0; i < 4; ++i) {
          int nt = np + 2 * (g * 4 + i);
          v16h B = frag_b(wp, nt * 16 + lrow, DIMC, ks, lhi);  // global, L2-hot
          acc[i] = wmma16(A, B, acc[i]);
        }
      }
#pragma unroll
      for (int i = 0; i < 4; ++i) {
        int nt = np + 2 * (g * 4 + i);
        int gc = nt * 16 + lrow;
        float bias = pb[gc];
#pragma unroll
        for (int r = 0; r < 8; ++r) {
          int t = mt * 16 + lhi * 8 + r;
          if (t < NTOK)
            out[((size_t)b * NTOK + t) * DIMC + gc] = acc[i][r] + bias;
        }
      }
    }
  }
}

// ---------------------------------------------------------------------------
// Launcher
// ---------------------------------------------------------------------------
extern "C" void kernel_launch(void* const* d_in, const int* in_sizes, int n_in,
                              void* d_out, int out_size, void* d_ws, size_t ws_size,
                              hipStream_t stream) {
  (void)in_sizes; (void)n_in; (void)out_size; (void)ws_size;
  const float* x     = (const float*)d_in[0];
  const float* mask  = (const float*)d_in[1];
  const float* qkvw  = (const float*)d_in[2];
  const float* qkvb  = (const float*)d_in[3];
  const float* projw = (const float*)d_in[4];
  const float* projb = (const float*)d_in[5];
  const float* rpb   = (const float*)d_in[6];
  const int*   ridx  = (const int*)d_in[7];
  float* out = (float*)d_out;

  char* ws = (char*)d_ws;
  _Float16* wq = (_Float16*)ws;                                          // 384 KB
  _Float16* wp = (_Float16*)(ws + (size_t)3 * DIMC * DIMC * sizeof(_Float16));

  k_convert_weights<<<(3 * DIMC * DIMC + 255) / 256, 256, 0, stream>>>(qkvw, projw, wq, wp);
  k_fused_window_attn<<<NWIN, 256, 0, stream>>>(x, mask, wq, qkvb, wp, projb,
                                                rpb, ridx, out);
}
